// MambaSelectiveSSM_83863531421726
// MI455X (gfx1250) — matrix operand order repaired and checked by reading, not compiled, not run on hardware
//
#include <hip/hip_runtime.h>

// ---------------------------------------------------------------------------
// Mamba selective SSM forward for MI455X (gfx1250, wave32, WMMA).
//   S*:  one-shot f32 -> (bf16 hi, bf16 lo) split planes for all GEMM operands
//   G1:  xp    = x @ W_xproj^T                    (1024 x 288,  K=4096)
//   G2:  delta = softplus(xp[:,:256] @ W_dt^T+b)  (1024 x 4096, K=256)
//   K3:  h_new = exp(delta*A)*h + (delta*B)*x ;  y_pre = C.h_new + D*x
//   G3:  y     = y_pre @ W_out^T                  (1024 x 4096, K=4096)
//
// GEMMs use the 3xBF16 split (Xhi*Whi + Xhi*Wlo + Xlo*Whi, f32 accumulate):
// ~16 effective mantissa bits at ~2.7x the native V_WMMA_F32_16X16X4_F32 rate.
// Hi/lo planes are precomputed once per operand so the GEMM inner loop is pure
// bf16 fragment loads + v_wmma. W_out's two bf16 planes (64 MB) fit in the
// 192 MB L2, so B-tile reloads across M-blocks hit L2, not HBM.
//
// Fragment addressing uses 32-bit element offsets off the kernel-arg base so
// InferAddressSpaces selects global_load_b128 (saddr+voffset) instead of
// flat_load_b128 (which would tie up DScnt and force s_wait_xcnt chains).
// ---------------------------------------------------------------------------

typedef __attribute__((ext_vector_type(8)))  __bf16 v8bf;
typedef __attribute__((ext_vector_type(16))) __bf16 v16bf;
typedef __attribute__((ext_vector_type(8)))  float  v8f;
typedef __attribute__((ext_vector_type(4)))  float  v4f;

constexpr int kDModel = 4096;
constexpr int kDState = 16;
constexpr int kDtRank = 256;                    // D_MODEL / 16
constexpr int kBatch  = 1024;
constexpr int kXprojN = kDtRank + 2 * kDState;  // 288

__device__ __forceinline__ float softplus_f(float v) {
  return fmaxf(v, 0.0f) + log1pf(__expf(-fabsf(v)));
}

// Load a 16-element bf16 fragment: 8 bf16 at plane[off], 8 at plane[off+step].
// A-fragments use step=16 (K-groups {0..7, 16..23} per half), B uses step=16 too.
__device__ __forceinline__ v16bf load_frag(const __bf16* __restrict__ plane,
                                           int off, int step) {
  v8bf q0 = *(const v8bf*)(plane + off);
  v8bf q1 = *(const v8bf*)(plane + off + step);
  return __builtin_shufflevector(q0, q1, 0, 1, 2, 3, 4, 5, 6, 7,
                                 8, 9, 10, 11, 12, 13, 14, 15);
}

__device__ __forceinline__ v8f wmma3(v16bf ah, v16bf al, v16bf bh, v16bf bl, v8f c) {
  c = __builtin_amdgcn_wmma_f32_16x16x32_bf16(false, ah, false, bh,
                                              (short)0, c, false, false);
  c = __builtin_amdgcn_wmma_f32_16x16x32_bf16(false, ah, false, bl,
                                              (short)0, c, false, false);
  c = __builtin_amdgcn_wmma_f32_16x16x32_bf16(false, al, false, bh,
                                              (short)0, c, false, false);
  return c;
}

// ---------------------------------------------------------------------------
// Split kernels: f32 -> bf16 hi + bf16 residual lo.
// ---------------------------------------------------------------------------
__global__ __launch_bounds__(256) void split_kernel(
    const float* __restrict__ src, __bf16* __restrict__ hi,
    __bf16* __restrict__ lo, int n) {
  int i = blockIdx.x * 256 + threadIdx.x;
  if (i < n) {
    float x  = src[i];
    __bf16 h = (__bf16)x;
    hi[i]    = h;
    lo[i]    = (__bf16)(x - (float)h);
  }
}

// Split a column slice src[r, 0:cols] (row stride ld) into compact planes.
__global__ __launch_bounds__(256) void split_slice_kernel(
    const float* __restrict__ src, int ld, __bf16* __restrict__ hi,
    __bf16* __restrict__ lo, int rows, int cols) {
  int i = blockIdx.x * 256 + threadIdx.x;
  if (i < rows * cols) {
    int r    = i / cols;
    int c    = i - r * cols;
    float x  = src[r * ld + c];
    __bf16 h = (__bf16)x;
    hi[i]    = h;
    lo[i]    = (__bf16)(x - (float)h);
  }
}

// ---------------------------------------------------------------------------
// Y[M,N] = act(X[M,K] @ W[N,K]^T + bias); X,W given as bf16 hi/lo planes.
// Block: 256 threads = 8 waves (4 M x 2 N), wave tile 32x64 (2x4 WMMA tiles),
// block tile 128x128. Requires K % 32 == 0 (4096, 256, 4096 all qualify).
// All plane sizes < 2^31 elements, so 32-bit element offsets suffice.
// ---------------------------------------------------------------------------
template <bool HAS_BIAS, bool DO_SOFTPLUS>
__global__ __launch_bounds__(256) void gemm_bf16x3_kernel(
    const __bf16* __restrict__ Xhi, const __bf16* __restrict__ Xlo, int ldx,
    const __bf16* __restrict__ Whi, const __bf16* __restrict__ Wlo,
    const float* __restrict__ bias,
    float* __restrict__ Y, int ldy,
    int M, int N, int K) {
  const int tid  = threadIdx.x;
  const int wave = tid >> 5;
  const int lane = tid & 31;
  const int half = lane >> 4;   // K-group select within fragments
  const int lr   = lane & 15;   // row (A) / col (B,D) within 16-wide tile

  const int wm = wave >> 1;                     // 0..3
  const int wn = wave & 1;                      // 0..1
  const int m0 = blockIdx.y * 128 + wm * 32;
  const int n0 = blockIdx.x * 128 + wn * 64;

  v8f acc[2][4] = {};

  // A: lane covers row m0+mt*16+lr; K offsets {half*8+0..7, half*8+16..23}.
  int aoff[2];
#pragma unroll
  for (int mt = 0; mt < 2; ++mt) {
    int r    = m0 + mt * 16 + lr;
    aoff[mt] = ((r < M) ? r : 0) * ldx + half * 8;
  }
  // B: lane covers col n0+nt*16+lr (a row of W); K offsets {half*8+0..7, half*8+16..23}.
  int boff[4];
#pragma unroll
  for (int nt = 0; nt < 4; ++nt) {
    int c    = n0 + nt * 16 + lr;
    boff[nt] = ((c < N) ? c : 0) * K + half * 8;
  }
  // (OOB lanes clamp to row/col 0: their garbage lands only in lanes that the
  //  guarded epilogue never stores.)

  for (int k0 = 0; k0 < K; k0 += 32) {
    v16bf ah[2], al[2], bh[4], bl[4];
#pragma unroll
    for (int mt = 0; mt < 2; ++mt) {
      ah[mt] = load_frag(Xhi, aoff[mt], 16);
      al[mt] = load_frag(Xlo, aoff[mt], 16);
      aoff[mt] += 32;
    }
#pragma unroll
    for (int nt = 0; nt < 4; ++nt) {
      bh[nt] = load_frag(Whi, boff[nt], 16);
      bl[nt] = load_frag(Wlo, boff[nt], 16);
      boff[nt] += 32;
    }
#pragma unroll
    for (int mt = 0; mt < 2; ++mt)
#pragma unroll
      for (int nt = 0; nt < 4; ++nt)
        acc[mt][nt] = wmma3(ah[mt], al[mt], bh[nt], bl[nt], acc[mt][nt]);
  }

  // Epilogue: C/D layout — elem r -> row m0+mt*16+half*8+r, col n0+nt*16+lr.
#pragma unroll
  for (int mt = 0; mt < 2; ++mt)
#pragma unroll
    for (int nt = 0; nt < 4; ++nt) {
      int gc = n0 + nt * 16 + lr;
      if (gc >= N) continue;
      float bv = HAS_BIAS ? bias[gc] : 0.0f;
#pragma unroll
      for (int r = 0; r < 8; ++r) {
        int gr = m0 + mt * 16 + half * 8 + r;
        if (gr < M) {
          float v = acc[mt][nt][r] + bv;
          if (DO_SOFTPLUS) v = softplus_f(v);
          Y[(size_t)gr * ldy + gc] = v;
        }
      }
    }
}

// ---------------------------------------------------------------------------
// One thread per (b, d): streams h[b,d,0:16] (64B) in, h_new (64B) out.
// Bandwidth-bound: ~0.54 GB over h/h_new -> ~23 us floor at 23.3 TB/s.
// Emits y_pre directly as bf16 hi/lo planes for G3.
// ---------------------------------------------------------------------------
__global__ __launch_bounds__(256) void ssm_update_kernel(
    const float* __restrict__ x, const float* __restrict__ h,
    const float* __restrict__ xp, const float* __restrict__ delta,
    const float* __restrict__ A_log, const float* __restrict__ Dv,
    float* __restrict__ h_new, __bf16* __restrict__ ypre_hi,
    __bf16* __restrict__ ypre_lo) {
  __shared__ float sA[kDState], sB[kDState], sC[kDState];
  const int b   = blockIdx.x;
  const int tid = threadIdx.x;
  if (tid < kDState) {
    sA[tid] = -__expf(A_log[tid]);
    sB[tid] = xp[b * kXprojN + kDtRank + tid];
    sC[tid] = xp[b * kXprojN + kDtRank + kDState + tid];
  }
  __syncthreads();

  const int d   = blockIdx.y * blockDim.x + tid;
  const int bd  = b * kDModel + d;          // < 2^22
  const int bd4 = bd * (kDState / 4);       // v4f granules, < 2^24
  const float xv = x[bd];
  const float dl = delta[bd];

  float hv[kDState];
#pragma unroll
  for (int q = 0; q < 4; ++q) *(v4f*)(hv + 4 * q) = ((const v4f*)h)[bd4 + q];

  float y = 0.0f;
#pragma unroll
  for (int n = 0; n < kDState; ++n) {
    float a_bar = __expf(dl * sA[n]);  // exp(delta * A[n])
    float v     = fmaf(a_bar, hv[n], (dl * sB[n]) * xv);
    hv[n]       = v;
    y           = fmaf(sC[n], v, y);
  }

#pragma unroll
  for (int q = 0; q < 4; ++q) ((v4f*)h_new)[bd4 + q] = *(const v4f*)(hv + 4 * q);

  float yv    = fmaf(Dv[d], xv, y);
  __bf16 yh   = (__bf16)yv;
  ypre_hi[bd] = yh;
  ypre_lo[bd] = (__bf16)(yv - (float)yh);
}

extern "C" void kernel_launch(void* const* d_in, const int* in_sizes, int n_in,
                              void* d_out, int out_size, void* d_ws, size_t ws_size,
                              hipStream_t stream) {
  const float* x       = (const float*)d_in[0];
  const float* h       = (const float*)d_in[1];
  const float* W_xproj = (const float*)d_in[2];
  const float* W_dt    = (const float*)d_in[3];
  const float* b_dt    = (const float*)d_in[4];
  const float* A_log   = (const float*)d_in[5];
  const float* Dv      = (const float*)d_in[6];
  const float* W_out   = (const float*)d_in[7];

  float* y_out = (float*)d_out;                     // [1024*4096]
  float* h_new = y_out + (size_t)kBatch * kDModel;  // [1024*4096*16]

  // ---- workspace carve (256B aligned; total ~130 MB) ----
  char* wp   = (char*)d_ws;
  auto carve = [&](size_t bytes) -> char* {
    char* r = wp;
    wp += (bytes + 255) & ~(size_t)255;
    return r;
  };
  const size_t nX   = (size_t)kBatch * kDModel;    // 4.19M
  const size_t nXP  = (size_t)kBatch * kXprojN;    // 0.29M
  const size_t nWxp = (size_t)kXprojN * kDModel;   // 1.18M
  const size_t nWdt = (size_t)kDModel * kDtRank;   // 1.05M
  const size_t nDr  = (size_t)kBatch * kDtRank;    // 0.26M
  const size_t nWo  = (size_t)kDModel * kDModel;   // 16.8M

  float*  xp     = (float*)carve(nXP * 4);
  float*  delta  = (float*)carve(nX * 4);
  __bf16* x_hi   = (__bf16*)carve(nX * 2);
  __bf16* x_lo   = (__bf16*)carve(nX * 2);
  __bf16* wxp_hi = (__bf16*)carve(nWxp * 2);
  __bf16* wxp_lo = (__bf16*)carve(nWxp * 2);
  __bf16* wdt_hi = (__bf16*)carve(nWdt * 2);
  __bf16* wdt_lo = (__bf16*)carve(nWdt * 2);
  __bf16* dr_hi  = (__bf16*)carve(nDr * 2);
  __bf16* dr_lo  = (__bf16*)carve(nDr * 2);
  __bf16* yp_hi  = (__bf16*)carve(nX * 2);
  __bf16* yp_lo  = (__bf16*)carve(nX * 2);
  __bf16* wo_hi  = (__bf16*)carve(nWo * 2);
  __bf16* wo_lo  = (__bf16*)carve(nWo * 2);

  dim3 blk(256);

  // ---- one-shot operand splits ----
  split_kernel<<<(int)((nWo + 255) / 256), blk, 0, stream>>>(W_out, wo_hi, wo_lo, (int)nWo);
  split_kernel<<<(int)((nX + 255) / 256), blk, 0, stream>>>(x, x_hi, x_lo, (int)nX);
  split_kernel<<<(int)((nWxp + 255) / 256), blk, 0, stream>>>(W_xproj, wxp_hi, wxp_lo, (int)nWxp);
  split_kernel<<<(int)((nWdt + 255) / 256), blk, 0, stream>>>(W_dt, wdt_hi, wdt_lo, (int)nWdt);

  // G1: xp = x @ W_xproj^T   (M=1024, N=288, K=4096)
  gemm_bf16x3_kernel<false, false>
      <<<dim3((kXprojN + 127) / 128, kBatch / 128), blk, 0, stream>>>(
          x_hi, x_lo, kDModel, wxp_hi, wxp_lo, nullptr,
          xp, kXprojN, kBatch, kXprojN, kDModel);

  // delta_raw = xp[:, :256] -> bf16 planes
  split_slice_kernel<<<(int)((nDr + 255) / 256), blk, 0, stream>>>(
      xp, kXprojN, dr_hi, dr_lo, kBatch, kDtRank);

  // G2: delta = softplus(delta_raw @ W_dt^T + b_dt)  (M=1024, N=4096, K=256)
  gemm_bf16x3_kernel<true, true>
      <<<dim3(kDModel / 128, kBatch / 128), blk, 0, stream>>>(
          dr_hi, dr_lo, kDtRank, wdt_hi, wdt_lo, b_dt,
          delta, kDModel, kBatch, kDModel, kDtRank);

  // K3: state update + y_pre planes (bandwidth bound)
  ssm_update_kernel<<<dim3(kBatch, kDModel / 256), blk, 0, stream>>>(
      x, h, xp, delta, A_log, Dv, h_new, yp_hi, yp_lo);

  // G3: y = y_pre @ W_out^T   (M=1024, N=4096, K=4096) -> dominant GEMM
  gemm_bf16x3_kernel<false, false>
      <<<dim3(kDModel / 128, kBatch / 128), blk, 0, stream>>>(
          yp_hi, yp_lo, kDModel, wo_hi, wo_lo, nullptr,
          y_out, kDModel, kBatch, kDModel, kDModel);
}
